// Document_Encoder_GAE_38448547234392
// MI455X (gfx1250) — matrix operand order repaired and checked by reading, not compiled, not run on hardware
//
#include <hip/hip_runtime.h>

// ---------------------------------------------------------------------------
// Types for CDNA5 WMMA (wave32): v_wmma_f32_16x16x32_bf16
// ---------------------------------------------------------------------------
typedef __bf16 bf16_t;
typedef bf16_t v16bf __attribute__((ext_vector_type(16)));
typedef bf16_t v8bf  __attribute__((ext_vector_type(8)));
typedef float  v8f   __attribute__((ext_vector_type(8)));

union FragBF { v16bf v; v8bf h[2]; };

__device__ __forceinline__ bf16_t to_bf16(float f) {
  union { float f; unsigned u; } x; x.f = f;
  // round-to-nearest-even truncation to bf16
  unsigned r = x.u + 0x7FFFu + ((x.u >> 16) & 1u);
  unsigned short hi = (unsigned short)(r >> 16);
  bf16_t b;
  __builtin_memcpy(&b, &hi, 2);
  return b;
}

// ---------------------------------------------------------------------------
// Degree / normalization kernels
// ---------------------------------------------------------------------------
__global__ void deg_init_kernel(float* __restrict__ deg, int n) {
  int i = blockIdx.x * blockDim.x + threadIdx.x;
  if (i < n) deg[i] = 1.0f;               // self-loop contribution
}

__global__ void deg_accum_kernel(const int* __restrict__ col,
                                 float* __restrict__ deg, int e) {
  int i = blockIdx.x * blockDim.x + threadIdx.x;
  if (i < e) unsafeAtomicAdd(&deg[col[i]], 1.0f);
}

__global__ void deg_rsqrt_kernel(float* __restrict__ deg, int n) {
  int i = blockIdx.x * blockDim.x + threadIdx.x;
  if (i < n) deg[i] = rsqrtf(deg[i]);     // deg >= 1 always (self-loops)
}

// ---------------------------------------------------------------------------
// fp32 -> bf16 conversions
// ---------------------------------------------------------------------------
__global__ void f32_to_bf16_kernel(const float* __restrict__ in,
                                   bf16_t* __restrict__ out, long n) {
  long i = (long)blockIdx.x * blockDim.x + threadIdx.x;
  if (i < n) out[i] = to_bf16(in[i]);
}

__global__ void relu_to_bf16_kernel(const float* __restrict__ in,
                                    bf16_t* __restrict__ out, long n) {
  long i = (long)blockIdx.x * blockDim.x + threadIdx.x;
  if (i < n) out[i] = to_bf16(fmaxf(in[i], 0.0f));
}

// W is [K, N] row-major fp32; write Wt as [N, K] row-major bf16 so the WMMA
// B-operand (lane = n, 16 consecutive K per lane-half) loads contiguously.
__global__ void transpose_to_bf16_kernel(const float* __restrict__ W,
                                         bf16_t* __restrict__ Wt,
                                         int K, int N) {
  int idx = blockIdx.x * blockDim.x + threadIdx.x;
  if (idx >= K * N) return;
  int k = idx / N, n = idx % N;
  Wt[(size_t)n * K + k] = to_bf16(W[idx]);
}

// ---------------------------------------------------------------------------
// GEMM: C[M,N] = A[M,K](bf16, row-major) * Bt[N,K](bf16, N-major) , f32 acc
// 256 threads = 8 waves; wave w -> rows [blockIdx.x*128 + w*16, +16),
// block covers 64 columns (4 x 16x16 WMMA tiles per wave).
// ---------------------------------------------------------------------------
__global__ void __launch_bounds__(256)
gemm_bf16_wmma_kernel(const bf16_t* __restrict__ A,
                      const bf16_t* __restrict__ Bt,
                      float* __restrict__ C,
                      int M, int N, int K) {
  const int wave = threadIdx.x >> 5;
  const int lane = threadIdx.x & 31;
  const int m0   = blockIdx.x * 128 + wave * 16;
  if (m0 >= M) return;                    // wave-uniform: EXEC stays all-ones
  const int n0 = blockIdx.y * 64;
  const int ml = lane & 15;               // M index within tile (A); N (B/C)
  const int hf = lane >> 4;               // lane half selects K sub-range

  v8f acc[4] = {};

  const bf16_t* arow = A + (size_t)(m0 + ml) * K;
  for (int k0 = 0; k0 < K; k0 += 32) {
    // prefetch next A chunk (global_prefetch_b8)
    __builtin_prefetch(arow + k0 + 32, 0, 1);

    // A fragment 16x32: lanes 0-15 K=[k0+0..15], lanes 16-31 K=[k0+16..31]
    FragBF a;
    a.h[0] = *(const v8bf*)(arow + k0 + hf * 16);
    a.h[1] = *(const v8bf*)(arow + k0 + hf * 16 + 8);

#pragma unroll
    for (int t = 0; t < 4; ++t) {
      // B fragment 32x16: lane n = ml, K = k0 + hf*16 + [0..15], contiguous
      const bf16_t* bp = Bt + (size_t)(n0 + t * 16 + ml) * K + k0 + hf * 16;
      FragBF b;
      b.h[0] = *(const v8bf*)(bp);
      b.h[1] = *(const v8bf*)(bp + 8);
      acc[t] = __builtin_amdgcn_wmma_f32_16x16x32_bf16(
          false, a.v, false, b.v, (short)0, acc[t], false, false);
    }
  }

  // C/D layout: VGPR r holds M = r + 8*hf, N = n0 + t*16 + ml
#pragma unroll
  for (int t = 0; t < 4; ++t) {
#pragma unroll
    for (int r = 0; r < 8; ++r) {
      int m = m0 + r + 8 * hf;
      C[(size_t)m * N + (n0 + t * 16 + ml)] = acc[t][r];
    }
  }
}

// ---------------------------------------------------------------------------
// out[i,c] = dinv[i]^2 * XW[i,c] + bias[c]   (self-loop message + bias)
// ---------------------------------------------------------------------------
__global__ void init_selfloop_bias_kernel(const float* __restrict__ XW,
                                          const float* __restrict__ dinv,
                                          const float* __restrict__ bias,
                                          float* __restrict__ out,
                                          int C, long total) {
  long idx = (long)blockIdx.x * blockDim.x + threadIdx.x;
  if (idx >= total) return;
  long i = idx / C;
  int  c = (int)(idx - i * C);
  float d = dinv[i];
  out[idx] = d * d * XW[idx] + bias[c];
}

// ---------------------------------------------------------------------------
// Edge scatter: out[col] += dinv[row]*dinv[col] * XW[row]  (float4 + f32 atomics)
// ---------------------------------------------------------------------------
template <int C>
__global__ void __launch_bounds__(256)
scatter_edges_kernel(const float* __restrict__ XW,
                     const int* __restrict__ rows,
                     const int* __restrict__ cols,
                     const float* __restrict__ dinv,
                     float* __restrict__ out, int E) {
  constexpr int TPE = C / 4;              // threads per edge (float4 lanes)
  constexpr int EPB = 256 / TPE;          // edges per block
  int e = blockIdx.x * EPB + (int)(threadIdx.x / TPE);
  if (e >= E) return;
  int lane = threadIdx.x % TPE;
  int r = rows[e];
  int c = cols[e];
  float norm = dinv[r] * dinv[c];
  float4 v = ((const float4*)(XW + (size_t)r * C))[lane];
  float* dst = out + (size_t)c * C + lane * 4;
  unsafeAtomicAdd(dst + 0, v.x * norm);
  unsafeAtomicAdd(dst + 1, v.y * norm);
  unsafeAtomicAdd(dst + 2, v.z * norm);
  unsafeAtomicAdd(dst + 3, v.w * norm);
}

// ---------------------------------------------------------------------------
// Host-side launch
// ---------------------------------------------------------------------------
extern "C" void kernel_launch(void* const* d_in, const int* in_sizes, int n_in,
                              void* d_out, int out_size, void* d_ws, size_t ws_size,
                              hipStream_t stream) {
  const float* X  = (const float*)d_in[0];   // [N, 256]
  const int*   EI = (const int*)d_in[1];     // [2, E]
  const float* W1 = (const float*)d_in[2];   // [256, 256]
  const float* b1 = (const float*)d_in[3];   // [256]
  const float* W2 = (const float*)d_in[4];   // [256, 128]
  const float* b2 = (const float*)d_in[5];   // [128]
  float* out = (float*)d_out;                // [N, 128]
  (void)n_in; (void)out_size; (void)ws_size;

  const int IN_CH = 256, HID = 256, OUT = 128;
  const int N = in_sizes[0] / IN_CH;         // 20000 (multiple of 16)
  const int E = in_sizes[1] / 2;             // 320000
  const int* rows = EI;
  const int* cols = EI + E;

  // workspace layout
  char* ws = (char*)d_ws;
  size_t off = 0;
  auto carve = [&](size_t bytes) -> void* {
    void* p = ws + off;
    off = (off + bytes + 255) & ~(size_t)255;
    return p;
  };
  bf16_t* Xbf  = (bf16_t*)carve((size_t)N * IN_CH * sizeof(bf16_t)); // also Hbf
  float*  XW1  = (float*) carve((size_t)N * HID * sizeof(float));
  float*  Hacc = (float*) carve((size_t)N * HID * sizeof(float));
  float*  XW2  = (float*) carve((size_t)N * OUT * sizeof(float));
  bf16_t* W1t  = (bf16_t*)carve((size_t)IN_CH * HID * sizeof(bf16_t));
  bf16_t* W2t  = (bf16_t*)carve((size_t)HID * OUT * sizeof(bf16_t));
  float*  dinv = (float*) carve((size_t)N * sizeof(float));

  const int T = 256;

  // 1) symmetric normalization coefficients
  deg_init_kernel<<<(N + T - 1) / T, T, 0, stream>>>(dinv, N);
  deg_accum_kernel<<<(E + T - 1) / T, T, 0, stream>>>(cols, dinv, E);
  deg_rsqrt_kernel<<<(N + T - 1) / T, T, 0, stream>>>(dinv, N);

  // 2) precision conversion / weight transpose for WMMA B-operand
  long nx = (long)N * IN_CH;
  f32_to_bf16_kernel<<<(unsigned)((nx + T - 1) / T), T, 0, stream>>>(X, Xbf, nx);
  transpose_to_bf16_kernel<<<(IN_CH * HID + T - 1) / T, T, 0, stream>>>(W1, W1t, IN_CH, HID);
  transpose_to_bf16_kernel<<<(HID * OUT + T - 1) / T, T, 0, stream>>>(W2, W2t, HID, OUT);

  // 3) layer 1: GEMM (WMMA bf16) -> self-loop+bias init -> edge scatter -> relu
  dim3 g1((unsigned)((N / 16 + 7) / 8), HID / 64);
  gemm_bf16_wmma_kernel<<<g1, 256, 0, stream>>>(Xbf, W1t, XW1, N, HID, IN_CH);

  long nh = (long)N * HID;
  init_selfloop_bias_kernel<<<(unsigned)((nh + T - 1) / T), T, 0, stream>>>(
      XW1, dinv, b1, Hacc, HID, nh);
  scatter_edges_kernel<256><<<(E + 3) / 4, 256, 0, stream>>>(
      XW1, rows, cols, dinv, Hacc, E);
  relu_to_bf16_kernel<<<(unsigned)((nh + T - 1) / T), T, 0, stream>>>(Hacc, Xbf, nh);

  // 4) layer 2: GEMM (WMMA bf16) -> self-loop+bias init -> edge scatter
  dim3 g2((unsigned)((N / 16 + 7) / 8), OUT / 64);
  gemm_bf16_wmma_kernel<<<g2, 256, 0, stream>>>(Xbf, W2t, XW2, N, OUT, HID);

  long no = (long)N * OUT;
  init_selfloop_bias_kernel<<<(unsigned)((no + T - 1) / T), T, 0, stream>>>(
      XW2, dinv, b2, out, OUT, no);
  scatter_edges_kernel<128><<<(E + 7) / 8, 256, 0, stream>>>(
      XW2, rows, cols, dinv, out, E);
}
